// Attention_45844480917562
// MI455X (gfx1250) — matrix operand order repaired and checked by reading, not compiled, not run on hardware
//
#include <hip/hip_runtime.h>
#include <hip/hip_bf16.h>

// ---------------- problem constants ----------------
#define DIM        4096
#define NHEADS     32
#define NKVH       8
#define HD         128
#define NREP       4
#define TOTB       32          // 16 + 16
#define B0         16
#define SP0        2048
#define SP1        1024
#define CLEN       2080
#define QKVC       6144        // 4096 q + 1024 k + 1024 v
#define SPLITK     4
#define NSPLIT     8           // attention kv splits

typedef __attribute__((ext_vector_type(16))) _Float16 v16h;
typedef __attribute__((ext_vector_type(8)))  float    v8f;

// convert a float4 into 4 consecutive f16 lanes of a v16h fragment
__device__ __forceinline__ void cvt4(v16h& r, int o, float4 v) {
    r[o + 0] = (_Float16)v.x;
    r[o + 1] = (_Float16)v.y;
    r[o + 2] = (_Float16)v.z;
    r[o + 3] = (_Float16)v.w;
}

// ---------------------------------------------------------------------------
// GEMM: Cpart[s][32][LDC] (cols coloff..coloff+N) = X[32,K] @ W[N,K]^T over
// the K-range owned by split s = blockIdx.y.  One wave per 16-col tile, two
// 16x16 WMMA accumulators cover M=32.  f32 weights/activations converted to
// f16 in-register; f32 accumulate (V_WMMA_F32_16X16X32_F16).  LDC is a
// template constant so the store epilogue is pure shift/add addressing.
// ---------------------------------------------------------------------------
template <int LDC>
__global__ void gemm_splitk(const float* __restrict__ X,
                            const float* __restrict__ W,
                            float* __restrict__ Cpart,
                            int N, int K, int coloff) {
    const int wave = threadIdx.x >> 5;
    const int lane = threadIdx.x & 31;
    const int tile = blockIdx.x * (blockDim.x >> 5) + wave;
    if (tile * 16 >= N) return;               // wave-uniform exit (EXEC stays all-1s)

    const int n0  = tile * 16;
    const int s   = blockIdx.y;
    const int kch = K >> 2;                   // SPLITK == 4
    const int k0  = s * kch;
    const int nl  = lane & 15;                // column within tile / M row base
    const int sel = lane >> 4;                // which K-half this lane owns

    // B operand: row (n0+nl) of W, K sel*8+(0..7) and 16+sel*8+(0..7).
    const float* wrow  = W + (size_t)(n0 + nl) * K + k0 + sel * 8;
    // A operand rows m = nl and m = nl+16
    const float* a0row = X + (size_t)nl        * K + k0 + sel * 8;
    const float* a1row = X + (size_t)(nl + 16) * K + k0 + sel * 8;

    v8f c0 = {}; v8f c1 = {};

#pragma unroll 2
    for (int kk = 0; kk < kch; kk += 32) {
        // prefetch next k-tile of the weight stream into near cache
        // (locality 3 -> WGP-scope global_prefetch_b8: fills all cache levels,
        //  consumed by this same wave one iteration later)
        __builtin_prefetch(wrow + kk + 32, 0, 3);

        v16h bf, af0, af1;
        {   // B: elems 0..7 = K sel*8+(0..7), elems 8..15 = K 16+sel*8+(0..7)
            const float4* p = (const float4*)(wrow + kk);
            cvt4(bf, 0,  p[0]); cvt4(bf, 4,  p[1]);
            cvt4(bf, 8,  p[4]); cvt4(bf, 12, p[5]);
        }
        {   // A row m=nl : elems 0..7 = K sel*8+(0..7), elems 8..15 = K 16+sel*8+(0..7)
            const float* p = a0row + kk;
            cvt4(af0, 0,  *(const float4*)(p));
            cvt4(af0, 4,  *(const float4*)(p + 4));
            cvt4(af0, 8,  *(const float4*)(p + 16));
            cvt4(af0, 12, *(const float4*)(p + 20));
        }
        {   // A row m=nl+16
            const float* p = a1row + kk;
            cvt4(af1, 0,  *(const float4*)(p));
            cvt4(af1, 4,  *(const float4*)(p + 4));
            cvt4(af1, 8,  *(const float4*)(p + 16));
            cvt4(af1, 12, *(const float4*)(p + 20));
        }
        c0 = __builtin_amdgcn_wmma_f32_16x16x32_f16(false, af0, false, bf,
                                                    (short)0, c0, false, false);
        c1 = __builtin_amdgcn_wmma_f32_16x16x32_f16(false, af1, false, bf,
                                                    (short)0, c1, false, false);
    }

    // C layout: VGPR g -> M = g + (lane>=16 ? 8 : 0), N = lane&15
    float* out = Cpart + s * (TOTB * LDC) + coloff + n0 + nl;
    const int mh = sel * 8;
#pragma unroll
    for (int g = 0; g < 8; g++) {
        out[(mh + g)      * LDC] = c0[g];
        out[(mh + g + 16) * LDC] = c1[g];
    }
}

// ---------------------------------------------------------------------------
// Reduce the 4 K-split partials of the QKV GEMM and apply RoPE to q & k.
// One thread per (row, even/odd column pair).  32 * 3072 threads.
// ---------------------------------------------------------------------------
__global__ void rope_reduce_qkv(const float* __restrict__ part,
                                float* __restrict__ qkv,
                                const float* __restrict__ fcos,
                                const float* __restrict__ fsin) {
    const int tid = blockIdx.x * blockDim.x + threadIdx.x;
    if (tid >= TOTB * (QKVC / 2)) return;
    const int b   = tid / (QKVC / 2);
    const int col = (tid % (QKVC / 2)) * 2;

    float e0 = 0.f, e1 = 0.f;
#pragma unroll
    for (int s = 0; s < SPLITK; s++) {
        const float* p = part + s * (TOTB * QKVC) + b * QKVC + col;
        e0 += p[0]; e1 += p[1];
    }
    if (col < DIM + NKVH * HD) {              // q (0..4095) and k (4096..5119) get RoPE
        const int pos = (b < B0) ? SP0 : SP1;
        const int i   = (col & (HD - 1)) >> 1;
        const float c = fcos[pos * (HD / 2) + i];
        const float s = fsin[pos * (HD / 2) + i];
        const float r0 = e0 * c - e1 * s;
        const float r1 = e0 * s + e1 * c;
        e0 = r0; e1 = r1;
    }
    qkv[b * QKVC + col]     = e0;
    qkv[b * QKVC + col + 1] = e1;
}

// ---------------------------------------------------------------------------
// Flash-decode partial: one wave per (batch, kv-head, split).  Each lane owns
// 4 head dims; K/V rows streamed as float4 (coalesced 512B rows, read once).
// 4 GQA q-heads share each K/V row.  Online softmax in registers; partial
// (acc[128], m, l) written to workspace.
// ---------------------------------------------------------------------------
__global__ void attn_partial(const float* __restrict__ qkv,
                             const float* __restrict__ ck0, const float* __restrict__ cv0,
                             const float* __restrict__ ck1, const float* __restrict__ cv1,
                             float* __restrict__ part) {
    const int wid  = (blockIdx.x * blockDim.x + threadIdx.x) >> 5;
    const int lane = threadIdx.x & 31;
    const int s    = wid & (NSPLIT - 1);
    const int kvh  = (wid >> 3) & (NKVH - 1);
    const int b    = wid >> 6;

    const int start_pos = (b < B0) ? SP0 : SP1;
    const int kv_len    = start_pos + 1;
    const int chunk     = (kv_len + NSPLIT - 1) / NSPLIT;
    const int p0        = s * chunk;
    int       p1        = p0 + chunk; if (p1 > kv_len) p1 = kv_len;

    const float* ck; const float* cv; int bl = b;
    if (b < B0) { ck = ck0; cv = cv0; } else { ck = ck1; cv = cv1; bl = b - B0; }
    const size_t cbase = (size_t)bl * CLEN * (NKVH * HD) + (size_t)kvh * HD;

    const float* qrow = qkv + b * QKVC;
    const int    hq   = kvh * NREP;
    float4 qv[NREP];
#pragma unroll
    for (int g = 0; g < NREP; g++)
        qv[g] = *(const float4*)(qrow + (hq + g) * HD + lane * 4);

    const float* knew = qrow + DIM + kvh * HD;
    const float* vnew = qrow + DIM + NKVH * HD + kvh * HD;

    float  m[NREP], l[NREP]; float4 acc[NREP];
#pragma unroll
    for (int g = 0; g < NREP; g++) {
        m[g] = -1e30f; l[g] = 0.f; acc[g] = make_float4(0.f, 0.f, 0.f, 0.f);
    }
    const float scale = 0.08838834764831845f;   // 1/sqrt(128)

    for (int p = p0; p < p1; p++) {
        const float* kp; const float* vp;
        if (p < start_pos) {
            kp = ck + cbase + (size_t)p * (NKVH * HD);
            vp = cv + cbase + (size_t)p * (NKVH * HD);
        } else { kp = knew; vp = vnew; }

        // prefetch next position's K/V rows into near cache while computing
        if (p + 1 < start_pos) {
            __builtin_prefetch(kp + NKVH * HD + lane * 4, 0, 3);
            __builtin_prefetch(vp + NKVH * HD + lane * 4, 0, 3);
        }

        const float4 kv4 = *(const float4*)(kp + lane * 4);
        float sc[NREP];
#pragma unroll
        for (int g = 0; g < NREP; g++) {
            float d = qv[g].x * kv4.x + qv[g].y * kv4.y +
                      qv[g].z * kv4.z + qv[g].w * kv4.w;
#pragma unroll
            for (int off = 16; off > 0; off >>= 1) d += __shfl_xor(d, off, 32);
            sc[g] = d * scale;
        }
        const float4 vv = *(const float4*)(vp + lane * 4);
#pragma unroll
        for (int g = 0; g < NREP; g++) {
            const float nm = fmaxf(m[g], sc[g]);
            const float cf = __expf(m[g] - nm);
            const float pw = __expf(sc[g] - nm);
            m[g] = nm;
            l[g] = l[g] * cf + pw;
            acc[g].x = acc[g].x * cf + pw * vv.x;
            acc[g].y = acc[g].y * cf + pw * vv.y;
            acc[g].z = acc[g].z * cf + pw * vv.z;
            acc[g].w = acc[g].w * cf + pw * vv.w;
        }
    }
#pragma unroll
    for (int g = 0; g < NREP; g++) {
        float* ob = part + (((s * TOTB + b) * NHEADS) + hq + g) * 132;
        *(float4*)(ob + lane * 4) = acc[g];
        if (lane == 0) { ob[128] = m[g]; ob[129] = l[g]; }   // wave-uniform m,l
    }
}

// Combine the NSPLIT flash-decode partials.  One thread per (b, h, d).
__global__ void attn_reduce(const float* __restrict__ part,
                            float* __restrict__ out) {
    const int tid = blockIdx.x * blockDim.x + threadIdx.x;
    if (tid >= TOTB * NHEADS * HD) return;
    const int d = tid & (HD - 1);
    const int h = (tid >> 7) & (NHEADS - 1);
    const int b = tid >> 12;

    float mv[NSPLIT], lv[NSPLIT], M = -1e30f;
#pragma unroll
    for (int s = 0; s < NSPLIT; s++) {
        const float* ob = part + (((s * TOTB + b) * NHEADS) + h) * 132;
        mv[s] = ob[128]; lv[s] = ob[129];
        M = fmaxf(M, mv[s]);
    }
    float L = 0.f, A = 0.f;
#pragma unroll
    for (int s = 0; s < NSPLIT; s++) {
        const float w = __expf(mv[s] - M);
        L += lv[s] * w;
        A += part[(((s * TOTB + b) * NHEADS) + h) * 132 + d] * w;
    }
    out[b * DIM + h * HD + d] = A / L;
}

// Sum the 4 K-split partials of the output projection into d_out.
__global__ void sum_splitk(const float* __restrict__ part,
                           float* __restrict__ out, int n) {
    const int i = blockIdx.x * blockDim.x + threadIdx.x;
    if (i >= n) return;
    out[i] = part[i] + part[n + i] + part[2 * n + i] + part[3 * n + i];
}

// ---------------------------------------------------------------------------
extern "C" void kernel_launch(void* const* d_in, const int* in_sizes, int n_in,
                              void* d_out, int out_size, void* d_ws, size_t ws_size,
                              hipStream_t stream) {
    const float* x   = (const float*)d_in[0];
    const float* wq  = (const float*)d_in[1];
    const float* wk  = (const float*)d_in[2];
    const float* wv  = (const float*)d_in[3];
    const float* wo  = (const float*)d_in[4];
    const float* ck0 = (const float*)d_in[5];
    const float* cv0 = (const float*)d_in[6];
    const float* ck1 = (const float*)d_in[7];
    const float* cv1 = (const float*)d_in[8];
    const float* fc  = (const float*)d_in[9];
    const float* fs  = (const float*)d_in[10];
    float* out = (float*)d_out;

    // workspace layout (floats)
    float* ws        = (float*)d_ws;
    float* qkv_part  = ws;                                   // 4*32*6144
    float* qkv       = qkv_part + (size_t)SPLITK * TOTB * QKVC;
    float* attn_part = qkv      + (size_t)TOTB * QKVC;       // 8*32*32*132
    float* attn_out  = attn_part + (size_t)NSPLIT * TOTB * NHEADS * 132;
    float* out_part  = attn_out + (size_t)TOTB * DIM;        // 4*32*4096

    // 1) QKV projection (split-K=4 WMMA GEMMs into one partial buffer)
    gemm_splitk<QKVC><<<dim3(DIM / 16 / 4, SPLITK), 128, 0, stream>>>(
        x, wq, qkv_part, DIM, DIM, 0);
    gemm_splitk<QKVC><<<dim3((NKVH * HD) / 16 / 4, SPLITK), 128, 0, stream>>>(
        x, wk, qkv_part, NKVH * HD, DIM, DIM);
    gemm_splitk<QKVC><<<dim3((NKVH * HD) / 16 / 4, SPLITK), 128, 0, stream>>>(
        x, wv, qkv_part, NKVH * HD, DIM, DIM + NKVH * HD);

    // 2) reduce split-K partials + RoPE(q,k)
    {
        const int n = TOTB * (QKVC / 2);
        rope_reduce_qkv<<<(n + 255) / 256, 256, 0, stream>>>(qkv_part, qkv, fc, fs);
    }

    // 3) flash-decode partials: 32 b * 8 kvh * 8 splits waves, 8 waves/block
    attn_partial<<<(TOTB * NKVH * NSPLIT) / 8, 256, 0, stream>>>(
        qkv, ck0, cv0, ck1, cv1, attn_part);

    // 4) combine splits
    {
        const int n = TOTB * NHEADS * HD;
        attn_reduce<<<(n + 255) / 256, 256, 0, stream>>>(attn_part, attn_out);
    }

    // 5) output projection (split-K=4 WMMA GEMM)
    gemm_splitk<DIM><<<dim3(DIM / 16 / 4, SPLITK), 128, 0, stream>>>(
        attn_out, wo, out_part, DIM, DIM, 0);

    // 6) final split-K reduction into d_out
    {
        const int n = TOTB * DIM;
        sum_splitk<<<(n + 255) / 256, 256, 0, stream>>>(out_part, out, n);
    }
}
